// HPUMLAImpl_68616397521564
// MI455X (gfx1250) — matrix operand order repaired and checked by reading, not compiled, not run on hardware
//
#include <hip/hip_runtime.h>
#include <hip/hip_bf16.h>

typedef __bf16 bf16;
typedef __attribute__((ext_vector_type(16))) __bf16 v16bf;
typedef __attribute__((ext_vector_type(8)))  float  v8f;
typedef unsigned int u32x4 __attribute__((ext_vector_type(4)));
typedef int          i32x8 __attribute__((ext_vector_type(8)));
typedef int          i32x4 __attribute__((ext_vector_type(4)));

#define B_      32
#define S_      4096
#define H_      16
#define QC_     1536
#define LORA_   512
#define ROPE_   64
#define NOPE_   128
#define VH_     128
#define DM_     2048
#define KD_     576            // LORA_ + ROPE_ combined latent-key width
#define CHUNK_  512
#define NCHUNK_ 8
#define SCALE_  0.07216878364870323f   // 1/sqrt(192)

// ---------------- CDNA5 data-mover paths (probe-guarded) ----------------
#if defined(__gfx1250__) && __has_builtin(__builtin_amdgcn_tensor_load_to_lds) && \
    __has_builtin(__builtin_amdgcn_s_wait_tensorcnt)
#define TDM_LDS 1
#else
#define TDM_LDS 0
#endif

#if defined(__AMDGCN__) && __has_builtin(__builtin_amdgcn_global_load_async_to_lds_b128)
#define ASYNC_LDS 1
#else
#define ASYNC_LDS 0
#endif

#if ASYNC_LDS && __has_builtin(__builtin_amdgcn_s_wait_asynccnt)
#define WAIT_ASYNC_ALL() __builtin_amdgcn_s_wait_asynccnt(0)
#else
#define WAIT_ASYNC_ALL()
#endif

// Stage-pipeline waits: TDM issues 1 op per tile, async path issues 4.
#if TDM_LDS
#define WAIT_STAGE_ONE()  __builtin_amdgcn_s_wait_tensorcnt(1)
#define WAIT_STAGE_ZERO() __builtin_amdgcn_s_wait_tensorcnt(0)
#elif ASYNC_LDS
#define WAIT_STAGE_ONE()  __builtin_amdgcn_s_wait_asynccnt(4)
#define WAIT_STAGE_ZERO() __builtin_amdgcn_s_wait_asynccnt(0)
#else
#define WAIT_STAGE_ONE()
#define WAIT_STAGE_ZERO()
#endif

typedef int v4i_vs __attribute__((vector_size(16)));
typedef __attribute__((address_space(1))) v4i_vs* gbl_v4i_p;   // clang prints AS1 as __device__
typedef __attribute__((address_space(3))) v4i_vs* lds_v4i_p;

__device__ __forceinline__ void async_cp16(float* lds_dst, const float* gsrc) {
#if ASYNC_LDS
    __builtin_amdgcn_global_load_async_to_lds_b128(
        (gbl_v4i_p)(float*)gsrc,
        (lds_v4i_p)lds_dst, 0, 0);
#else
    *(float4*)lds_dst = *(const float4*)gsrc;
#endif
}

// Stage one 16-position x 32-feature f32 tile (row stride `stride` f32) into
// LDS, row-major [pos][feat]. TDM path: single TENSOR_LOAD_TO_LDS with a 2D
// D# (data_size=4B, tile 32x16, dim0_stride=stride). Wave-uniform args.
__device__ __forceinline__ void stage_tile(float* buf, const float* src, int stride) {
#if TDM_LDS
    const unsigned lds_off = (unsigned)(uintptr_t)(lds_v4i_p)buf;   // LDS byte offset
    const unsigned long long ga = (unsigned long long)(uintptr_t)src;
    u32x4 g0;
    g0[0] = 1u;                                            // count=1, user mode
    g0[1] = lds_off;                                       // lds_addr
    g0[2] = (unsigned)ga;                                  // global_addr[31:0]
    g0[3] = (unsigned)((ga >> 32) & 0x1FFFFFFu) | (2u << 30); // addr[56:32] | type=2
    i32x8 g1;
    g1[0] = 2 << 16;                                       // data_size = 4 bytes
    g1[1] = 32 << 16;                                      // tensor_dim0 = 32
    g1[2] = 16 << 16;                                      // tensor_dim1 = 16
    g1[3] = 32 << 16;                                      // tile_dim0 = 32
    g1[4] = 16;                                            // tile_dim1 = 16 (tile_dim2=0)
    g1[5] = stride;                                        // tensor_dim0_stride[31:0]
    g1[6] = 0; g1[7] = 0;                                  // stride hi / dim1_stride
    const i32x4 z4 = {0, 0, 0, 0};
    const i32x8 z8 = {0, 0, 0, 0, 0, 0, 0, 0};
    __builtin_amdgcn_tensor_load_to_lds(g0, g1, z4, z4, z8, 0);
#else
    const int lane = threadIdx.x & 31;
#pragma unroll
    for (int j = 0; j < 4; ++j) {
        const int q  = j * 32 + lane;   // float4 index (0..127)
        const int r  = q >> 3;          // row 0..15
        const int c4 = (q & 7) * 4;     // feature within tile
        async_cp16(buf + r * 32 + c4, src + (size_t)r * stride + c4);
    }
#endif
}

// =====================  Kernel 1: Q projection + RoPE  =====================
// grid (H, B), 128 threads. Writes q (bf16, pre-scaled by SCALE) [B][H][576]
// and the patched new key row kcomb (f32) [B][576].
__global__ __launch_bounds__(128) void mla_qproj_kernel(
    const float* __restrict__ q_c, const float* __restrict__ k_c,
    const float* __restrict__ k_pe,
    const float* __restrict__ W_UQ, const float* __restrict__ W_UK,
    const float* __restrict__ W_QR, const int* __restrict__ posp,
    bf16* __restrict__ qws, float* __restrict__ kcomb)
{
    __shared__ __align__(16) float qc_s[QC_];
    __shared__ float qt_s[NOPE_];
    __shared__ float qpe_s[ROPE_];

    const int b = blockIdx.y, h = blockIdx.x, t = threadIdx.x;
    // async global->LDS copy of the q_c row (384 float4, 3 per thread)
#pragma unroll
    for (int j = 0; j < 3; ++j) {
        const int o = (t + 128 * j) * 4;
        async_cp16(qc_s + o, q_c + (size_t)b * QC_ + o);
    }
    WAIT_ASYNC_ALL();
    __syncthreads();

    // q_t[h][t] = sum_d q_c[d] * W_UQ[d][h][t]
    {
        float s = 0.f;
        for (int d = 0; d < QC_; ++d)
            s += qc_s[d] * W_UQ[((size_t)d * H_ + h) * NOPE_ + t];
        qt_s[t] = s;
    }
    // pre-rope q_pe
    if (t < ROPE_) {
        float s = 0.f;
        for (int d = 0; d < QC_; ++d)
            s += qc_s[d] * W_QR[(size_t)d * (H_ * ROPE_) + h * ROPE_ + t];
        qpe_s[t] = s;
    }
    __syncthreads();

    bf16* qrow = qws + ((size_t)b * H_ + h) * KD_;
    // q_nope[h][l] = sum_n q_t[n] * W_UK[l][h][n]
    for (int l = t; l < LORA_; l += 128) {
        float s = 0.f;
#pragma unroll 4
        for (int n = 0; n < NOPE_; ++n)
            s += qt_s[n] * W_UK[((size_t)l * H_ + h) * NOPE_ + n];
        qrow[l] = (bf16)(s * SCALE_);
    }
    // RoPE on q_pe (half = 32)
    if (t < ROPE_ / 2) {
        const int p = posp[b];
        const float inv = __powf(10000.f, -(float)t / (float)(ROPE_ / 2));
        const float f = (float)p * inv;
        const float cs = __cosf(f), sn = __sinf(f);
        const float x1 = qpe_s[t], x2 = qpe_s[t + ROPE_ / 2];
        qrow[LORA_ + t]            = (bf16)((x1 * cs - x2 * sn) * SCALE_);
        qrow[LORA_ + ROPE_/2 + t]  = (bf16)((x2 * cs + x1 * sn) * SCALE_);
    }
    // new key row (only h==0 blocks)
    if (h == 0) {
        for (int i = t; i < LORA_; i += 128)
            kcomb[(size_t)b * KD_ + i] = k_c[(size_t)b * LORA_ + i];
        if (t < ROPE_ / 2) {
            const int p = posp[b];
            const float inv = __powf(10000.f, -(float)t / (float)(ROPE_ / 2));
            const float f = (float)p * inv;
            const float cs = __cosf(f), sn = __sinf(f);
            const float x1 = k_pe[(size_t)b * ROPE_ + t];
            const float x2 = k_pe[(size_t)b * ROPE_ + ROPE_/2 + t];
            kcomb[(size_t)b * KD_ + LORA_ + t]           = x1 * cs - x2 * sn;
            kcomb[(size_t)b * KD_ + LORA_ + ROPE_/2 + t] = x2 * cs + x1 * sn;
        }
    }
}

// =====================  Kernel 2: split-K flash decode  =====================
// grid (NCHUNK, B), 256 threads = 8 wave32.
__global__ __launch_bounds__(256) void mla_attn_kernel(
    const float* __restrict__ kv, const float* __restrict__ kpe,
    const bf16* __restrict__ qws, const float* __restrict__ kcomb,
    const int* __restrict__ posp,
    float* __restrict__ o_part, float* __restrict__ m_part, float* __restrict__ l_part)
{
    __shared__ float sc[H_ * CHUNK_];                    // 32 KB scores
    __shared__ __align__(16) unsigned char st[32768];    // staging, then P(bf16)+m/l

    const int b = blockIdx.y, c = blockIdx.x;
    const int tid = threadIdx.x, wave = tid >> 5, lane = tid & 31;
    const int pos = posp[b];
    const int s0 = c * CHUNK_;

    float* buf0 = (float*)st + wave * 1024;   // per-wave double buffer, 2x(16x32) f32
    float* buf1 = buf0 + 512;

    const int mrow = lane & 15;               // head (A-matrix row)
    const int mb   = (lane < 16) ? 0 : 8;     // WMMA half-wave M offset (accumulator rows)
    const bf16* qrow = qws + (size_t)b * H_ * KD_;
    const float* krow = kcomb + (size_t)b * KD_;

    // ---- pass 1: score tiles, S = Q(16x576) . K^T(576x16), bf16 WMMA ----
    for (int t = 0; t < 4; ++t) {
        const int ts0 = (wave * 4 + t) * 16;  // local s-tile base
        const int srow0 = s0 + ts0;           // global first position of tile
        const int pr = pos - srow0;           // patched row, if 0..15 (wave-uniform)
        v8f acc = {0.f,0.f,0.f,0.f,0.f,0.f,0.f,0.f};

        // source of k-chunk kc for this tile (16 rows, 32 features)
        auto tile_src = [&](int kc) -> const float* {
            return (kc < 16) ? kv  + ((size_t)b * S_ + srow0) * LORA_ + kc * 32
                             : kpe + ((size_t)b * S_ + srow0) * ROPE_ + (kc - 16) * 32;
        };
        stage_tile(buf0, tile_src(0), LORA_);
        for (int kc = 0; kc < 18; ++kc) {
            float* cur = (kc & 1) ? buf1 : buf0;
            float* nxt = (kc & 1) ? buf0 : buf1;
            if (kc < 17) {
                stage_tile(nxt, tile_src(kc + 1), (kc + 1 < 16) ? LORA_ : ROPE_);
                WAIT_STAGE_ONE();     // previous tile's DMA complete (in-order)
            } else {
                WAIT_STAGE_ZERO();
            }
            // patch the freshly-written key row (wave-uniform branch, EXEC stays full)
            if (pr >= 0 && pr < 16)
                cur[pr * 32 + lane] = krow[kc * 32 + lane];
            // A fragment (q, bf16), contiguous k order k = 16*(lane/16) + s
            v16bf a;
            const bf16* p0 = qrow + mrow * KD_ + kc * 32 + (lane >> 4) * 16;
#pragma unroll
            for (int e = 0; e < 8; ++e) { a[e] = p0[e]; a[e + 8] = p0[8 + e]; }
            // B fragment: B[k][n] = keytile[n][k], cvt f32->bf16
            v16bf bb;
            const float* brow = cur + (lane & 15) * 32 + (lane >> 4) * 16;
#pragma unroll
            for (int i = 0; i < 16; ++i) bb[i] = (bf16)brow[i];
            acc = __builtin_amdgcn_wmma_f32_16x16x32_bf16(false, a, false, bb,
                                                          (short)0, acc, false, false);
        }
        // C/D layout: lane holds col N=lane%16, rows mb..mb+7
#pragma unroll
        for (int r = 0; r < 8; ++r)
            sc[(mb + r) * CHUNK_ + ts0 + (lane & 15)] = acc[r];
    }
    __syncthreads();

    // ---- pass 2: softmax stats + P (bf16 in LDS), then O = P . KV ----
    bf16*  P  = (bf16*)st;            // 16x512 bf16 = 16 KB (aliases staging)
    float* ml = (float*)(st + 16384); // [0..15]=m, [16..31]=l
    {
        const int h = tid >> 4, sl = tid & 15;
        float mx = -3.0e38f;
        for (int i = 0; i < 32; ++i) mx = fmaxf(mx, sc[h * CHUNK_ + sl * 32 + i]);
#pragma unroll
        for (int d = 8; d >= 1; d >>= 1) mx = fmaxf(mx, __shfl_xor(mx, d, 32));
        if (sl == 0) ml[h] = mx;
        __syncthreads();
        const float m = ml[h];
        float sum = 0.f;
        for (int i = 0; i < 32; ++i) {
            const float p = __expf(sc[h * CHUNK_ + sl * 32 + i] - m);
            P[h * CHUNK_ + sl * 32 + i] = (bf16)p;
            sum += p;
        }
#pragma unroll
        for (int d = 8; d >= 1; d >>= 1) sum += __shfl_xor(sum, d, 32);
        if (sl == 0) ml[16 + h] = sum;
    }
    __syncthreads();

    // O partial: [16 heads x 512 lora] = P[16x512] . KV[512x512]; wave owns 64 cols
    v8f oacc[4];
    {
        const v8f z = {0.f,0.f,0.f,0.f,0.f,0.f,0.f,0.f};
#pragma unroll
        for (int ct = 0; ct < 4; ++ct) oacc[ct] = z;
    }
    for (int kc = 0; kc < 16; ++kc) {          // K-chunks of 32 positions
        v16bf a;
        const bf16* p0 = P + mrow * CHUNK_ + kc * 32 + (lane >> 4) * 16;
#pragma unroll
        for (int e = 0; e < 8; ++e) { a[e] = p0[e]; a[e + 8] = p0[8 + e]; }
#pragma unroll
        for (int ct = 0; ct < 4; ++ct) {
            const int n0 = wave * 64 + ct * 16;
            const int n  = n0 + (lane & 15);
            const int kb = kc * 32 + (lane >> 4) * 16;
            v16bf bb;
#pragma unroll
            for (int i = 0; i < 16; ++i) {
                const int sg = s0 + kb + i;
                // flat branchless select between cache row and patched new row
                const uintptr_t pa = (uintptr_t)(kv + ((size_t)b * S_ + sg) * LORA_ + n);
                const uintptr_t pb = (uintptr_t)(krow + n);
                const float v = *(const float*)((sg == pos) ? pb : pa);
                bb[i] = (bf16)v;
            }
            oacc[ct] = __builtin_amdgcn_wmma_f32_16x16x32_bf16(false, a, false, bb,
                                                               (short)0, oacc[ct], false, false);
        }
    }
    float* op = o_part + ((size_t)b * NCHUNK_ + c) * H_ * LORA_;
#pragma unroll
    for (int ct = 0; ct < 4; ++ct)
#pragma unroll
        for (int r = 0; r < 8; ++r)
            op[(mb + r) * LORA_ + wave * 64 + ct * 16 + (lane & 15)] = oacc[ct][r];
    if (tid < H_) {
        m_part[((size_t)b * NCHUNK_ + c) * H_ + tid] = ml[tid];
        l_part[((size_t)b * NCHUNK_ + c) * H_ + tid] = ml[16 + tid];
    }
}

// ============  Kernel 3: merge partials + W_UV + W_O projection  ============
// grid (DM/256, B), 256 threads.
__global__ __launch_bounds__(256) void mla_out_kernel(
    const float* __restrict__ o_part, const float* __restrict__ m_part,
    const float* __restrict__ l_part,
    const float* __restrict__ W_UV, const float* __restrict__ W_O,
    float* __restrict__ out)
{
    __shared__ float om[H_ * LORA_];       // 32 KB merged, softmax-normalized O
    __shared__ float vsh[H_ * VH_];        // 8 KB
    __shared__ float wc[NCHUNK_ * H_];

    const int b = blockIdx.y, j = blockIdx.x, tid = threadIdx.x;

    if (tid < H_) {
        const int h = tid;
        float M = -3.0e38f;
        for (int c = 0; c < NCHUNK_; ++c)
            M = fmaxf(M, m_part[((size_t)b * NCHUNK_ + c) * H_ + h]);
        float den = 0.f;
        for (int c = 0; c < NCHUNK_; ++c)
            den += __expf(m_part[((size_t)b * NCHUNK_ + c) * H_ + h] - M) *
                   l_part[((size_t)b * NCHUNK_ + c) * H_ + h];
        const float rden = 1.f / den;
        for (int c = 0; c < NCHUNK_; ++c)
            wc[c * H_ + h] = __expf(m_part[((size_t)b * NCHUNK_ + c) * H_ + h] - M) * rden;
    }
    __syncthreads();

    for (int idx = tid; idx < H_ * LORA_; idx += 256) {
        const int h = idx >> 9, l = idx & (LORA_ - 1);
        float s = 0.f;
#pragma unroll
        for (int c = 0; c < NCHUNK_; ++c)
            s += wc[c * H_ + h] * o_part[(((size_t)b * NCHUNK_ + c) * H_ + h) * LORA_ + l];
        om[idx] = s;
    }
    __syncthreads();

    // v[h][vh] = sum_l om[h][l] * W_UV[l][h][vh]
    for (int idx = tid; idx < H_ * VH_; idx += 256) {
        const int h = idx >> 7, vh = idx & (VH_ - 1);
        float s = 0.f;
        for (int l = 0; l < LORA_; ++l)
            s += om[h * LORA_ + l] * W_UV[((size_t)l * H_ + h) * VH_ + vh];
        vsh[idx] = s;   // matches v.reshape(H*VH) ordering
    }
    __syncthreads();

    const int dm = j * 256 + tid;
    float s = 0.f;
    for (int k = 0; k < H_ * VH_; ++k)
        s += vsh[k] * W_O[(size_t)k * DM_ + dm];
    out[(size_t)b * DM_ + dm] = s;
}

// ================================  launch  ================================
extern "C" void kernel_launch(void* const* d_in, const int* in_sizes, int n_in,
                              void* d_out, int out_size, void* d_ws, size_t ws_size,
                              hipStream_t stream)
{
    (void)in_sizes; (void)n_in; (void)out_size; (void)ws_size;
    const float* q_c  = (const float*)d_in[0];
    const float* k_c  = (const float*)d_in[1];
    const float* k_pe = (const float*)d_in[2];
    const float* kv   = (const float*)d_in[3];
    const float* kpec = (const float*)d_in[4];
    const float* W_UQ = (const float*)d_in[5];
    const float* W_UK = (const float*)d_in[6];
    const float* W_QR = (const float*)d_in[7];
    const float* W_UV = (const float*)d_in[8];
    const float* W_O  = (const float*)d_in[9];
    const int*   pos  = (const int*)d_in[10];

    char* w = (char*)d_ws;
    auto carve = [&](size_t bytes) -> char* {
        char* p = w; w += (bytes + 255) & ~(size_t)255; return p;
    };
    bf16*  qws    = (bf16*) carve((size_t)B_ * H_ * KD_ * sizeof(bf16));
    float* kcomb  = (float*)carve((size_t)B_ * KD_ * sizeof(float));
    float* o_part = (float*)carve((size_t)B_ * NCHUNK_ * H_ * LORA_ * sizeof(float));
    float* m_part = (float*)carve((size_t)B_ * NCHUNK_ * H_ * sizeof(float));
    float* l_part = (float*)carve((size_t)B_ * NCHUNK_ * H_ * sizeof(float));

    mla_qproj_kernel<<<dim3(H_, B_), 128, 0, stream>>>(
        q_c, k_c, k_pe, W_UQ, W_UK, W_QR, pos, qws, kcomb);
    mla_attn_kernel<<<dim3(NCHUNK_, B_), 256, 0, stream>>>(
        kv, kpec, qws, kcomb, pos, o_part, m_part, l_part);
    mla_out_kernel<<<dim3(DM_ / 256, B_), 256, 0, stream>>>(
        o_part, m_part, l_part, W_UV, W_O, (float*)d_out);
}
